// GraphAttentionTransformerOC20_16183436771987
// MI455X (gfx1250) — matrix operand order repaired and checked by reading, not compiled, not run on hardware
//
#include <hip/hip_runtime.h>
#include <math.h>

// ---------------- problem constants ----------------
#define N_    6000
#define E_    96000
#define B_    64
#define L_    2
#define C0_   256
#define C1_   128
#define NB_   128
#define H_    8
#define MLP0_ 768
#define MLP1_ 384
#define F_    512
#define NW_   640           // C0 + 3*C1
#define CUTOFF_ 6.0f
#define INV_AVG_DEG   (1.0f/23.395238876342773f)
#define INV_AVG_NODES (1.0f/77.81317f)

typedef __bf16 bf16;
typedef __attribute__((ext_vector_type(16))) __bf16 v16bf;
typedef __attribute__((ext_vector_type(8)))  float  v8f;
typedef int v4i_ __attribute__((ext_vector_type(4)));

#if __has_builtin(__builtin_amdgcn_global_load_async_to_lds_b128)
#define HAS_ASYNC_LDS 1
#else
#define HAS_ASYNC_LDS 0
#endif

// ---------------- small device helpers ----------------
__device__ __forceinline__ unsigned short f2bf(float f) {
  unsigned int u = __float_as_uint(f);
  u += 0x7fffu + ((u >> 16) & 1u);           // round-to-nearest-even
  return (unsigned short)(u >> 16);
}
__device__ __forceinline__ float silu_f(float x) { return x / (1.f + __expf(-x)); }
__device__ __forceinline__ float sigm_f(float x) { return 1.f / (1.f + __expf(-x)); }

__device__ __forceinline__ void atomicMaxF(float* a, float v) {
  unsigned int* u = (unsigned int*)a;
  unsigned int old = *u;
  while (__uint_as_float(old) < v) {
    unsigned int assumed = old;
    old = atomicCAS(u, assumed, __float_as_uint(v));
    if (old == assumed) break;
  }
}

// block-wide sum for 256-thread blocks (8 wave32), broadcasts result
__device__ __forceinline__ float blockSum256(float v) {
  __shared__ float sh[8];
  int lane = threadIdx.x & 31, w = threadIdx.x >> 5;
#pragma unroll
  for (int o = 16; o > 0; o >>= 1) v += __shfl_xor(v, o, 32);
  if (lane == 0) sh[w] = v;
  __syncthreads();
  if (w == 0) {
    float t = (lane < 8) ? sh[lane] : 0.f;
#pragma unroll
    for (int o = 4; o > 0; o >>= 1) t += __shfl_xor(t, o, 32);
    if (lane == 0) sh[0] = t;
  }
  __syncthreads();
  float r = sh[0];
  __syncthreads();
  return r;
}

// ---------------- WMMA bf16 GEMM ----------------
// C[M,Nn] = act(A[M,K](f32,lda) @ B[K,Nn](bf16, K-pair packed) + bias) (+= old C if ADD)
// Bp packing: uint Bp[K/2][Nn]; Bp[p][n] = bf16(B[2p][n]) | bf16(B[2p+1][n])<<16
// -> a GEMM LDS tile row is a straight 512B copy of global memory (async-LDS friendly)
#define BM 64
#define BN 128
#define BK 32

union FragAB { v16bf v; unsigned int u[8]; };

template<int ACT, int ADD>
__global__ __launch_bounds__(256)
void gemm_bf16_kernel(const float* __restrict__ A, int lda,
                      const unsigned int* __restrict__ Bp,
                      const float* __restrict__ bias,
                      float* __restrict__ C, int M, int Nn, int K)
{
  __shared__ __align__(16) unsigned short Abuf[BM * BK];   // row-major [64][32] bf16
  __shared__ __align__(16) unsigned int   Bbuf[(BK / 2) * BN]; // [16 kpair][128 n]
  const int tid  = threadIdx.x;
  const int lane = tid & 31;
  const int wid  = tid >> 5;
  const int wm   = wid >> 2;        // 0..1 : 32-row strip
  const int wn   = wid & 3;         // 0..3 : 32-col strip
  const int row0 = blockIdx.y * BM;
  const int col0 = blockIdx.x * BN;

  v8f acc[4] = {};

  const unsigned int* Au = (const unsigned int*)Abuf;  // row*16 + kpair

  // per-thread staging coordinates
  const int ar  = tid >> 3;            // A: row 0..31 (and +32)
  const int akq = (tid & 7) * 4;       // A: k offset 0..28 step 4
  const int bp  = tid >> 4;            // B: kpair row 0..15
  const int bn0 = (tid & 15) * 8;      // B: column (uint) 0..120 step 8
  const bool bvalid = (col0 + bn0) < Nn;     // Nn multiple of 16, bn0 of 8 => whole 8-uint run uniform

  for (int k0 = 0; k0 < K; k0 += BK) {
    // ---- stage B tile: straight copy of packed rows ----
    {
      unsigned int* ldst = Bbuf + bp * BN + bn0;
      const unsigned int* gsrc = Bp + (size_t)((k0 >> 1) + bp) * Nn + col0 + bn0;
      if (bvalid) {
#if HAS_ASYNC_LDS
        __builtin_amdgcn_global_load_async_to_lds_b128(
            (__attribute__((address_space(1))) v4i_*)gsrc,
            (__attribute__((address_space(3))) v4i_*)ldst, 0, 0);
        __builtin_amdgcn_global_load_async_to_lds_b128(
            (__attribute__((address_space(1))) v4i_*)(gsrc + 4),
            (__attribute__((address_space(3))) v4i_*)(ldst + 4), 0, 0);
#else
        uint4 q0 = *(const uint4*)gsrc;
        uint4 q1 = *(const uint4*)(gsrc + 4);
        *(uint4*)ldst = q0;
        *(uint4*)(ldst + 4) = q1;
#endif
      } else {
        uint4 z = {0u, 0u, 0u, 0u};
        *(uint4*)ldst = z;
        *(uint4*)(ldst + 4) = z;
      }
    }
    // ---- stage A tile (vectorized f32 -> bf16, branchless row clamp) ----
#pragma unroll
    for (int half = 0; half < 2; ++half) {
      int rr = ar + half * 32;
      int gr = row0 + rr;
      int cr = (gr < M) ? gr : (M - 1);
      float4 f4 = *(const float4*)(A + (size_t)cr * lda + k0 + akq);
      if (gr >= M) { f4.x = 0.f; f4.y = 0.f; f4.z = 0.f; f4.w = 0.f; }
      uint2 pk;
      pk.x = (unsigned)f2bf(f4.x) | ((unsigned)f2bf(f4.y) << 16);
      pk.y = (unsigned)f2bf(f4.z) | ((unsigned)f2bf(f4.w) << 16);
      *(uint2*)((unsigned int*)Abuf + rr * 16 + (akq >> 1)) = pk;
    }
    if (k0 + BK < K) {
      __builtin_prefetch(Bp + (size_t)(((k0 + BK) >> 1) + bp) * Nn + col0 + bn0, 0, 3);
    }
#if HAS_ASYNC_LDS
#if __has_builtin(__builtin_amdgcn_s_wait_asynccnt)
    __builtin_amdgcn_s_wait_asynccnt(0);
#else
    asm volatile("s_wait_asynccnt 0" ::: "memory");
#endif
#endif
    __syncthreads();

    // ---- build fragments (16-bit A 16x32 / B 32x16), contiguous k = 16*(lane/16) + elem for A and B ----
    FragAB a0, a1, b0, b1;
    {
      int arow = lane & 15;
      int akb  = (lane < 16) ? 0 : 8;       // K-pair offset (K 0..15 vs 16..31)
      const unsigned int* r0 = Au + (wm * 32 + arow) * 16;
      const unsigned int* r1 = Au + (wm * 32 + 16 + arow) * 16;
#pragma unroll
      for (int i = 0; i < 4; ++i) {
        a0.u[i]     = r0[akb + i];
        a0.u[i + 4] = r0[4 + akb + i];      // K = 8..15 / 24..31
        a1.u[i]     = r1[akb + i];
        a1.u[i + 4] = r1[4 + akb + i];
      }
      int bcol = lane & 15;
      int bkb  = (lane < 16) ? 0 : 8;       // K-pair offset (K 0..15 vs 16..31)
#pragma unroll
      for (int j = 0; j < 8; ++j) {
        b0.u[j] = Bbuf[(bkb + j) * BN + wn * 32 + bcol];
        b1.u[j] = Bbuf[(bkb + j) * BN + wn * 32 + 16 + bcol];
      }
    }
    acc[0] = __builtin_amdgcn_wmma_f32_16x16x32_bf16(false, a0.v, false, b0.v, (short)0, acc[0], false, false);
    acc[1] = __builtin_amdgcn_wmma_f32_16x16x32_bf16(false, a0.v, false, b1.v, (short)0, acc[1], false, false);
    acc[2] = __builtin_amdgcn_wmma_f32_16x16x32_bf16(false, a1.v, false, b0.v, (short)0, acc[2], false, false);
    acc[3] = __builtin_amdgcn_wmma_f32_16x16x32_bf16(false, a1.v, false, b1.v, (short)0, acc[3], false, false);
    __syncthreads();
  }

  // ---- epilogue: bias -> act -> accumulate ----
#pragma unroll
  for (int f = 0; f < 4; ++f) {
    int mi = wm * 32 + (f >> 1) * 16;
    int ni = wn * 32 + (f & 1) * 16;
    int cc = col0 + ni + (lane & 15);
    int rbase = row0 + mi + ((lane < 16) ? 0 : 8);
    if (cc < Nn) {
      float bv = bias ? bias[cc] : 0.f;
#pragma unroll
      for (int r = 0; r < 8; ++r) {
        int rr = rbase + r;
        if (rr < M) {
          float val = acc[f][r] + bv;
          if (ACT == 1) val = silu_f(val);
          if (ADD) val += C[(size_t)rr * Nn + cc];
          C[(size_t)rr * Nn + cc] = val;
        }
      }
    }
  }
}

// ---------------- elementwise / graph kernels ----------------
// pack f32 weights [Kt][Nn] -> uint [Kt/2][Nn], K-pair interleaved bf16
__global__ void pack_bf16_kernel(const float* __restrict__ w, unsigned int* __restrict__ out,
                                 int Kt, int Nn) {
  size_t idx = (size_t)blockIdx.x * 256 + threadIdx.x;
  size_t tot = (size_t)(Kt >> 1) * Nn;
  if (idx >= tot) return;
  int p = (int)(idx / Nn), n = (int)(idx % Nn);
  unsigned lo = f2bf(w[(size_t)(2 * p) * Nn + n]);
  unsigned hi = f2bf(w[(size_t)(2 * p + 1) * Nn + n]);
  out[idx] = lo | (hi << 16);
}

__global__ void fill_kernel(float* __restrict__ p, float v, size_t n) {
  size_t i = (size_t)blockIdx.x * 256 + threadIdx.x;
  if (i < n) p[i] = v;
}

__global__ void edge_vec_kernel(const float* __restrict__ pos, const int* __restrict__ src,
                                const int* __restrict__ dst, float* __restrict__ shv,
                                float* __restrict__ dist) {
  int e = blockIdx.x * 256 + threadIdx.x;
  if (e >= E_) return;
  int s = src[e], d = dst[e];
  float vx = pos[s * 3 + 0] - pos[d * 3 + 0];
  float vy = pos[s * 3 + 1] - pos[d * 3 + 1];
  float vz = pos[s * 3 + 2] - pos[d * 3 + 2];
  float dn = sqrtf(vx * vx + vy * vy + vz * vz + 1e-12f);
  float sc = 1.7320508075688772f / dn;     // sqrt(3)/dist
  shv[e] = vx * sc; shv[E_ + e] = vy * sc; shv[2 * (size_t)E_ + e] = vz * sc;
  dist[e] = dn;
}

__global__ void rbf_kernel(const float* __restrict__ dist, float* __restrict__ rbf) {
  size_t idx = (size_t)blockIdx.x * 256 + threadIdx.x;
  if (idx >= (size_t)E_ * NB_) return;
  int e = (int)(idx >> 7), j = (int)(idx & 127);
  float c = j * (CUTOFF_ / (NB_ - 1));
  float x = (dist[e] - c) * ((float)NB_ / CUTOFF_);
  rbf[idx] = __expf(-0.5f * x * x);
}

__global__ void node_init_kernel(const int* __restrict__ an, const int* __restrict__ tg,
                                 const float* __restrict__ atom_tab, const float* __restrict__ tag_tab,
                                 float* __restrict__ s) {
  int idx = blockIdx.x * 256 + threadIdx.x;
  if (idx >= N_ * C0_) return;
  int n = idx >> 8, c = idx & 255;
  s[idx] = atom_tab[an[n] * C0_ + c] + tag_tab[tg[n] * C0_ + c];
}

__global__ void scatter_deg_kernel(const float* __restrict__ dw, const int* __restrict__ dst,
                                   const float* __restrict__ shv, float* __restrict__ s,
                                   float* __restrict__ v) {
  size_t idx = (size_t)blockIdx.x * 256 + threadIdx.x;
  if (idx >= (size_t)E_ * 384) return;
  int e = (int)(idx / 384), c = (int)(idx % 384);
  float val = dw[(size_t)e * 384 + c] * INV_AVG_DEG;
  int d = dst[e];
  if (c < C0_) {
    atomicAdd(&s[(size_t)d * C0_ + c], val);
  } else {
    int cv = c - C0_;
    const size_t NV = (size_t)N_ * C1_;
#pragma unroll
    for (int i = 0; i < 3; ++i)
      atomicAdd(&v[i * NV + (size_t)d * C1_ + cv], val * shv[(size_t)i * E_ + e]);
  }
}

// equivariant layernorm: block per node, 256 threads
__global__ __launch_bounds__(256)
void eqln_kernel(const float* __restrict__ s, const float* __restrict__ v,
                 const float* __restrict__ w, const float* __restrict__ b,
                 const float* __restrict__ vw, float* __restrict__ sn, float* __restrict__ vn) {
  int n = blockIdx.x, t = threadIdx.x;
  const size_t NV = (size_t)N_ * C1_;
  float sv = s[(size_t)n * C0_ + t];
  float mu = blockSum256(sv) * (1.f / C0_);
  float d0 = sv - mu;
  float var = blockSum256(d0 * d0) * (1.f / C0_);
  float sd = sqrtf(var + 1e-5f);
  sn[(size_t)n * C0_ + t] = d0 / sd * w[t] + b[t];

  float vv0 = 0.f, vv1 = 0.f, vv2 = 0.f, sq = 0.f;
  if (t < C1_) {
    vv0 = v[0 * NV + (size_t)n * C1_ + t];
    vv1 = v[1 * NV + (size_t)n * C1_ + t];
    vv2 = v[2 * NV + (size_t)n * C1_ + t];
    sq = vv0 * vv0 + vv1 * vv1 + vv2 * vv2;
  }
  float vm = blockSum256(sq) * (1.f / C1_);
  float den = sqrtf(vm + 1e-5f);
  if (t < C1_) {
    float g = vw[t] / den;
    vn[0 * NV + (size_t)n * C1_ + t] = vv0 * g;
    vn[1 * NV + (size_t)n * C1_ + t] = vv1 * g;
    vn[2 * NV + (size_t)n * C1_ + t] = vv2 * g;
  }
}

// per-edge message (recomputed in both attention passes to save 245MB of ws)
__device__ __forceinline__ void edge_message(
    int e, int t, int sN, int dN,
    const float* __restrict__ fs, const float* __restrict__ gs,
    const float* __restrict__ fv, const float* __restrict__ gv,
    const float* __restrict__ wrad, const float* __restrict__ shv,
    float& ts, float tv[3]) {
  const size_t NV = (size_t)N_ * C1_;
  float ms = fs[(size_t)sN * C0_ + t] + gs[(size_t)dN * C0_ + t];
  const float* we = wrad + (size_t)e * NW_;
  ts = ms * we[t];
  tv[0] = tv[1] = tv[2] = 0.f;
  if (t < C1_) {
    float sh0 = shv[e], sh1 = shv[E_ + e], sh2 = shv[2 * (size_t)E_ + e];
    float mv0 = fv[0 * NV + (size_t)sN * C1_ + t] + gv[0 * NV + (size_t)dN * C1_ + t];
    float mv1 = fv[1 * NV + (size_t)sN * C1_ + t] + gv[1 * NV + (size_t)dN * C1_ + t];
    float mv2 = fv[2 * NV + (size_t)sN * C1_ + t] + gv[2 * NV + (size_t)dN * C1_ + t];
    float dot = mv0 * sh0 + mv1 * sh1 + mv2 * sh2;
    ts += dot * we[C0_ + t];
    float w3 = we[C0_ + C1_ + t], w4 = we[C0_ + 2 * C1_ + t];
    float m4 = ms * w4;
    tv[0] = mv0 * w3 + m4 * sh0;
    tv[1] = mv1 * w3 + m4 * sh1;
    tv[2] = mv2 * w3 + m4 * sh2;
  }
}

__global__ __launch_bounds__(256)
void edge_alpha_kernel(const int* __restrict__ src, const int* __restrict__ dst,
                       const float* __restrict__ fs, const float* __restrict__ gs,
                       const float* __restrict__ fv, const float* __restrict__ gv,
                       const float* __restrict__ wrad, const float* __restrict__ shv,
                       const float* __restrict__ alpha_w, float* __restrict__ alpha,
                       float* __restrict__ amax) {
  int e = blockIdx.x, t = threadIdx.x;
  int sN = src[e], dN = dst[e];
  float ts, tv[3];
  edge_message(e, t, sN, dN, fs, gs, fv, gv, wrad, shv, ts, tv);
  float lr = ts > 0.f ? ts : 0.2f * ts;          // leaky_relu(0.2)
  float av = lr * alpha_w[t];                     // alpha_w flat [H*H0]
#pragma unroll
  for (int o = 16; o > 0; o >>= 1) av += __shfl_xor(av, o, 32);
  int lane = t & 31, h = t >> 5;                  // head = wave id (32=H0 channels)
  if (lane == 0) {
    alpha[(size_t)e * H_ + h] = av;
    atomicMaxF(&amax[(size_t)dN * H_ + h], av);
  }
}

__global__ void edge_ea_kernel(const float* __restrict__ alpha, const int* __restrict__ dst,
                               const float* __restrict__ amax, float* __restrict__ ea,
                               float* __restrict__ easum) {
  int idx = blockIdx.x * 256 + threadIdx.x;
  if (idx >= E_ * H_) return;
  int e = idx >> 3, h = idx & 7;
  int d = dst[e];
  float x = __expf(alpha[idx] - amax[(size_t)d * H_ + h]);
  ea[idx] = x;
  atomicAdd(&easum[(size_t)d * H_ + h], x);
}

__global__ __launch_bounds__(256)
void edge_agg_kernel(const int* __restrict__ src, const int* __restrict__ dst,
                     const float* __restrict__ fs, const float* __restrict__ gs,
                     const float* __restrict__ fv, const float* __restrict__ gv,
                     const float* __restrict__ wrad, const float* __restrict__ shv,
                     const float* __restrict__ ea, const float* __restrict__ easum,
                     float* __restrict__ ags, float* __restrict__ agv) {
  int e = blockIdx.x, t = threadIdx.x;
  int sN = src[e], dN = dst[e];
  float ts, tv[3];
  edge_message(e, t, sN, dN, fs, gs, fv, gv, wrad, shv, ts, tv);
  int hs = t >> 5;                                // H0=32 channels per head
  float at_s = ea[(size_t)e * H_ + hs] / (easum[(size_t)dN * H_ + hs] + 1e-9f);
  atomicAdd(&ags[(size_t)dN * C0_ + t], ts * at_s);
  if (t < C1_) {
    int hv = t >> 4;                              // H1=16 channels per head
    float at_v = ea[(size_t)e * H_ + hv] / (easum[(size_t)dN * H_ + hv] + 1e-9f);
    const size_t NV = (size_t)N_ * C1_;
#pragma unroll
    for (int i = 0; i < 3; ++i)
      atomicAdd(&agv[i * NV + (size_t)dN * C1_ + t], tv[i] * at_v);
  }
}

__global__ void gate_kernel(float* __restrict__ h, float* __restrict__ hv) {
  size_t idx = (size_t)blockIdx.x * 256 + threadIdx.x;
  if (idx >= (size_t)N_ * (MLP0_ + MLP1_)) return;
  int n = (int)(idx / (MLP0_ + MLP1_)), c = (int)(idx % (MLP0_ + MLP1_));
  if (c < MLP0_) {
    h[idx] = silu_f(h[idx]);
  } else {
    int cg = c - MLP0_;
    float g = sigm_f(h[idx]);
    const size_t NH = (size_t)N_ * MLP1_;
#pragma unroll
    for (int i = 0; i < 3; ++i) hv[i * NH + (size_t)n * MLP1_ + cg] *= g;
  }
}

__global__ __launch_bounds__(256)
void layernorm_kernel(const float* __restrict__ f, const float* __restrict__ w,
                      const float* __restrict__ b, float* __restrict__ out) {
  int n = blockIdx.x, t = threadIdx.x;
  float x0 = f[(size_t)n * F_ + t], x1 = f[(size_t)n * F_ + 256 + t];
  float mu = blockSum256(x0 + x1) * (1.f / F_);
  float d0 = x0 - mu, d1 = x1 - mu;
  float var = blockSum256(d0 * d0 + d1 * d1) * (1.f / F_);
  float sd = sqrtf(var + 1e-5f);
  out[(size_t)n * F_ + t]       = d0 / sd * w[t] + b[t];
  out[(size_t)n * F_ + 256 + t] = d1 / sd * w[256 + t] + b[256 + t];
}

__global__ __launch_bounds__(256)
void head_kernel(const float* __restrict__ g1, const float* __restrict__ w2,
                 const float* __restrict__ b2, const int* __restrict__ batch,
                 float* __restrict__ out) {
  int n = blockIdx.x, t = threadIdx.x;
  float d = g1[(size_t)n * F_ + t] * w2[t] + g1[(size_t)n * F_ + 256 + t] * w2[256 + t];
  float s = blockSum256(d);
  if (t == 0) atomicAdd(&out[batch[n]], (s + b2[0]) * INV_AVG_NODES);
}

// ---------------- host orchestration ----------------
static void gemm(hipStream_t st, const float* A, int lda, const unsigned int* Bp,
                 const float* bias, float* C, int M, int Nn, int K, int act, int add) {
  dim3 grid((Nn + BN - 1) / BN, (M + BM - 1) / BM);
  dim3 blk(256);
  if (add)      gemm_bf16_kernel<0, 1><<<grid, blk, 0, st>>>(A, lda, Bp, bias, C, M, Nn, K);
  else if (act) gemm_bf16_kernel<1, 0><<<grid, blk, 0, st>>>(A, lda, Bp, bias, C, M, Nn, K);
  else          gemm_bf16_kernel<0, 0><<<grid, blk, 0, st>>>(A, lda, Bp, bias, C, M, Nn, K);
}

extern "C" void kernel_launch(void* const* d_in, const int* in_sizes, int n_in,
                              void* d_out, int out_size, void* d_ws, size_t ws_size,
                              hipStream_t stream) {
  (void)in_sizes; (void)n_in; (void)out_size; (void)ws_size;
  const float* pos = (const float*)d_in[0];
  const int* an    = (const int*)d_in[1];
  const int* tgs   = (const int*)d_in[2];
  const int* batch = (const int*)d_in[3];
  const int* eidx  = (const int*)d_in[4];
  const int* src = eidx;
  const int* dst = eidx + E_;
  const float* P[49];
  for (int i = 0; i < 49; ++i) P[i] = (const float*)d_in[i];

  // bump allocator over d_ws
  size_t off = 0;
  auto alloc = [&](size_t bytes) -> void* {
    void* p = (char*)d_ws + off;
    off += (bytes + 255) & ~(size_t)255;
    return p;
  };
  auto allocF = [&](size_t n) -> float* { return (float*)alloc(n * sizeof(float)); };
  auto allocU = [&](size_t n) -> unsigned int* { return (unsigned int*)alloc(n * sizeof(unsigned int)); };

  // packed bf16 weight copies (uint = 2 bf16, K-pair interleaved); sizes = K*N/2 uints
  unsigned int* b_deg1  = allocU((size_t)NB_ * 64 / 2);
  unsigned int* b_deg2  = allocU((size_t)64 * 64 / 2);
  unsigned int* b_deg3  = allocU((size_t)64 * 384 / 2);
  unsigned int* b_srcs  = allocU((size_t)L_ * C0_ * C0_ / 2);
  unsigned int* b_srcv  = allocU((size_t)L_ * C1_ * C1_ / 2);
  unsigned int* b_dsts  = allocU((size_t)L_ * C0_ * C0_ / 2);
  unsigned int* b_dstv  = allocU((size_t)L_ * C1_ * C1_ / 2);
  unsigned int* b_rad1  = allocU((size_t)L_ * NB_ * 64 / 2);
  unsigned int* b_rad2  = allocU((size_t)L_ * 64 * 64 / 2);
  unsigned int* b_rad3  = allocU((size_t)L_ * 64 * NW_ / 2);
  unsigned int* b_projs = allocU((size_t)L_ * C0_ * C0_ / 2);
  unsigned int* b_projv = allocU((size_t)L_ * C1_ * C1_ / 2);
  unsigned int* b_ffn1s = allocU((size_t)L_ * C0_ * (MLP0_ + MLP1_) / 2);
  unsigned int* b_ffn1v = allocU((size_t)L_ * C1_ * MLP1_ / 2);
  unsigned int* b_ffn2s = allocU((size_t)L_ * MLP0_ * C0_ / 2);
  unsigned int* b_ffn2v = allocU((size_t)L_ * MLP1_ * C1_ / 2);
  unsigned int* b_final = allocU((size_t)C0_ * F_ / 2);
  unsigned int* b_head1 = allocU((size_t)F_ * F_ / 2);

  // per-layer packed strides (uints)
  const size_t st_ss   = (size_t)C0_ * C0_ / 2;
  const size_t st_vv   = (size_t)C1_ * C1_ / 2;
  const size_t st_r1   = (size_t)NB_ * 64 / 2;
  const size_t st_r2   = (size_t)64 * 64 / 2;
  const size_t st_r3   = (size_t)64 * NW_ / 2;
  const size_t st_f1s  = (size_t)C0_ * (MLP0_ + MLP1_) / 2;
  const size_t st_f1v  = (size_t)C1_ * MLP1_ / 2;
  const size_t st_f2s  = (size_t)MLP0_ * C0_ / 2;
  const size_t st_f2v  = (size_t)MLP1_ * C1_ / 2;

  // f32 activations / intermediates
  float* shv   = allocF(3ull * E_);
  float* distb = allocF(E_);
  float* rbf   = allocF((size_t)E_ * NB_);
  float* s_    = allocF((size_t)N_ * C0_);
  float* v_    = allocF(3ull * N_ * C1_);
  float* sn    = allocF((size_t)N_ * C0_);
  float* vn    = allocF(3ull * N_ * C1_);
  float* fs    = allocF((size_t)N_ * C0_);
  float* gs    = allocF((size_t)N_ * C0_);
  float* fv    = allocF(3ull * N_ * C1_);
  float* gv    = allocF(3ull * N_ * C1_);
  float* rh1   = allocF((size_t)E_ * 64);
  float* rh2   = allocF((size_t)E_ * 64);
  float* wrad  = allocF((size_t)E_ * NW_);    // also used for deg output [E x 384]
  float* alpha = allocF((size_t)E_ * H_);
  float* ea    = allocF((size_t)E_ * H_);
  float* amax  = allocF((size_t)N_ * H_);
  float* easum = allocF((size_t)N_ * H_);
  float* ags   = allocF((size_t)N_ * C0_);
  float* agv   = allocF(3ull * N_ * C1_);
  float* hbuf  = allocF((size_t)N_ * (MLP0_ + MLP1_));
  float* hvbuf = allocF(3ull * N_ * MLP1_);
  float* fbuf  = allocF((size_t)N_ * F_);
  float* fnbuf = allocF((size_t)N_ * F_);
  float* g1buf = allocF((size_t)N_ * F_);

  auto packw = [&](const float* sp, unsigned int* dp, int Kt, int Nn) {
    size_t tot = (size_t)(Kt / 2) * Nn;
    pack_bf16_kernel<<<(unsigned)((tot + 255) / 256), 256, 0, stream>>>(sp, dp, Kt, Nn);
  };
  auto fillk = [&](float* p, float vval, size_t n) {
    fill_kernel<<<(unsigned)((n + 255) / 256), 256, 0, stream>>>(p, vval, n);
  };

  // ---- weight packing (f32 -> K-pair packed bf16); L-stacked weights pack in one shot
  // (per-layer K is even so pairs never straddle a layer boundary) ----
  packw(P[7],  b_deg1,  NB_, 64);
  packw(P[9],  b_deg2,  64, 64);
  packw(P[11], b_deg3,  64, 384);
  packw(P[16], b_srcs,  L_ * C0_, C0_);
  packw(P[18], b_srcv,  L_ * C1_, C1_);
  packw(P[19], b_dsts,  L_ * C0_, C0_);
  packw(P[21], b_dstv,  L_ * C1_, C1_);
  packw(P[22], b_rad1,  L_ * NB_, 64);
  packw(P[24], b_rad2,  L_ * 64, 64);
  packw(P[26], b_rad3,  L_ * 64, NW_);
  packw(P[29], b_projs, L_ * C0_, C0_);
  packw(P[31], b_projv, L_ * C1_, C1_);
  packw(P[35], b_ffn1s, L_ * C0_, MLP0_ + MLP1_);
  packw(P[37], b_ffn1v, L_ * C1_, MLP1_);
  packw(P[38], b_ffn2s, L_ * MLP0_, C0_);
  packw(P[40], b_ffn2v, L_ * MLP1_, C1_);
  packw(P[41], b_final, C0_, F_);
  packw(P[45], b_head1, F_, F_);

  // ---- edge geometry + RBF ----
  edge_vec_kernel<<<(E_ + 255) / 256, 256, 0, stream>>>(pos, src, dst, shv, distb);
  rbf_kernel<<<(unsigned)(((size_t)E_ * NB_ + 255) / 256), 256, 0, stream>>>(distb, rbf);

  // ---- node embedding + degree message ----
  node_init_kernel<<<(N_ * C0_ + 255) / 256, 256, 0, stream>>>(an, tgs, P[5], P[6], s_);
  fillk(v_, 0.f, 3ull * N_ * C1_);
  gemm(stream, rbf, NB_, b_deg1, P[8],  rh1,  E_, 64,  NB_, 1, 0);
  gemm(stream, rh1, 64,  b_deg2, P[10], rh2,  E_, 64,  64,  1, 0);
  gemm(stream, rh2, 64,  b_deg3, P[12], wrad, E_, 384, 64,  0, 0);
  scatter_deg_kernel<<<(unsigned)(((size_t)E_ * 384 + 255) / 256), 256, 0, stream>>>(wrad, dst, shv, s_, v_);

  // ---- message-passing layers ----
  for (int i = 0; i < L_; ++i) {
    eqln_kernel<<<N_, 256, 0, stream>>>(s_, v_, P[13] + i * C0_, P[14] + i * C0_, P[15] + i * C1_, sn, vn);
    gemm(stream, sn, C0_, b_srcs + i * st_ss, P[17] + i * C0_, fs, N_, C0_, C0_, 0, 0);
    gemm(stream, sn, C0_, b_dsts + i * st_ss, P[20] + i * C0_, gs, N_, C0_, C0_, 0, 0);
    gemm(stream, vn, C1_, b_srcv + i * st_vv, nullptr, fv, 3 * N_, C1_, C1_, 0, 0);
    gemm(stream, vn, C1_, b_dstv + i * st_vv, nullptr, gv, 3 * N_, C1_, C1_, 0, 0);

    gemm(stream, rbf, NB_, b_rad1 + i * st_r1, P[23] + i * 64, rh1, E_, 64, NB_, 1, 0);
    gemm(stream, rh1, 64,  b_rad2 + i * st_r2, P[25] + i * 64, rh2, E_, 64, 64,  1, 0);
    gemm(stream, rh2, 64,  b_rad3 + i * st_r3, P[27] + i * NW_, wrad, E_, NW_, 64, 0, 0);

    fillk(amax, -3.0e38f, (size_t)N_ * H_);
    fillk(easum, 0.f,     (size_t)N_ * H_);
    fillk(ags,   0.f,     (size_t)N_ * C0_);
    fillk(agv,   0.f,     3ull * N_ * C1_);

    edge_alpha_kernel<<<E_, 256, 0, stream>>>(src, dst, fs, gs, fv, gv, wrad, shv,
                                              P[28] + i * H_ * 32, alpha, amax);
    edge_ea_kernel<<<(E_ * H_ + 255) / 256, 256, 0, stream>>>(alpha, dst, amax, ea, easum);
    edge_agg_kernel<<<E_, 256, 0, stream>>>(src, dst, fs, gs, fv, gv, wrad, shv, ea, easum, ags, agv);

    gemm(stream, ags, C0_, b_projs + i * st_ss, P[30] + i * C0_, s_, N_, C0_, C0_, 0, 1);
    gemm(stream, agv, C1_, b_projv + i * st_vv, nullptr, v_, 3 * N_, C1_, C1_, 0, 1);

    eqln_kernel<<<N_, 256, 0, stream>>>(s_, v_, P[32] + i * C0_, P[33] + i * C0_, P[34] + i * C1_, sn, vn);
    gemm(stream, sn, C0_, b_ffn1s + i * st_f1s, P[36] + i * (MLP0_ + MLP1_),
         hbuf, N_, MLP0_ + MLP1_, C0_, 0, 0);
    gemm(stream, vn, C1_, b_ffn1v + i * st_f1v, nullptr, hvbuf, 3 * N_, MLP1_, C1_, 0, 0);
    gate_kernel<<<(unsigned)(((size_t)N_ * (MLP0_ + MLP1_) + 255) / 256), 256, 0, stream>>>(hbuf, hvbuf);
    gemm(stream, hbuf, MLP0_ + MLP1_, b_ffn2s + i * st_f2s, P[39] + i * C0_, s_, N_, C0_, MLP0_, 0, 1);
    gemm(stream, hvbuf, MLP1_, b_ffn2v + i * st_f2v, nullptr, v_, 3 * N_, C1_, MLP1_, 0, 1);
  }

  // ---- readout head ----
  gemm(stream, s_, C0_, b_final, P[42], fbuf, N_, F_, C0_, 0, 0);
  layernorm_kernel<<<N_, 256, 0, stream>>>(fbuf, P[43], P[44], fnbuf);
  gemm(stream, fnbuf, F_, b_head1, P[46], g1buf, N_, F_, F_, 1, 0);
  fillk((float*)d_out, 0.f, (size_t)B_);
  head_kernel<<<N_, 256, 0, stream>>>(g1buf, P[47], P[48], batch, (float*)d_out);
}
